// PCGraphConv_30356828848591
// MI455X (gfx1250) — compile-verified, matrix operand order checked by reading, not run on hardware
//
#include <hip/hip_runtime.h>
#include <hip/hip_bf16.h>

// ---------------- problem constants (match reference) ----------------
#define NV     2000      // vertices
#define NB     32        // batch
#define KPAD   2048      // padded K (and padded vertex dim for dense W)
#define NSENS  784       // sensory vertices (clamped)
#define TSTEPS 5
#define LRV    0.1f

typedef __attribute__((ext_vector_type(16))) _Float16 v16h;
typedef __attribute__((ext_vector_type(8)))  _Float16 v8h;
typedef __attribute__((ext_vector_type(8)))  float    v8f;

// ---------------- setup kernels ----------------

// zero dense f32 W accumulator (2048*2048 floats) as float4
__global__ __launch_bounds__(256) void k_zero4(float4* __restrict__ p) {
    p[(size_t)blockIdx.x * 256 + threadIdx.x] = make_float4(0.f, 0.f, 0.f, 0.f);
}

// scatter edges into dense f32 W: W[dst][src] += w  (duplicates accumulate)
__global__ __launch_bounds__(256) void k_scatter(const int* __restrict__ src,
                                                 const int* __restrict__ dst,
                                                 const float* __restrict__ w,
                                                 float* __restrict__ Wf, int nE) {
    int e = blockIdx.x * 256 + threadIdx.x;
    if (e < nE) {
        atomicAdd(&Wf[(size_t)dst[e] * KPAD + src[e]], w[e]);
    }
}

// convert f32 W -> f16 Wh (row-major) and WhT (transposed), LDS-tiled so both
// writes are coalesced. Grid = (2048/32)^2 blocks, 256 threads.
__global__ __launch_bounds__(256) void k_convert(const float* __restrict__ Wf,
                                                 _Float16* __restrict__ Wh,
                                                 _Float16* __restrict__ WhT) {
    __shared__ _Float16 tile[32][33];
    int tx = threadIdx.x & 31, ty = threadIdx.x >> 5;      // 32x8 thread tile
    int bi = blockIdx.x & 63,  bj = blockIdx.x >> 6;       // 64x64 grid of 32x32 tiles
    int i0 = bi * 32, j0 = bj * 32;
    #pragma unroll
    for (int r = ty; r < 32; r += 8) {
        float v = Wf[(size_t)(i0 + r) * KPAD + j0 + tx];
        _Float16 h = (_Float16)v;
        Wh[(size_t)(i0 + r) * KPAD + j0 + tx] = h;          // coalesced
        tile[r][tx] = h;
    }
    __syncthreads();
    #pragma unroll
    for (int r = ty; r < 32; r += 8) {
        WhT[(size_t)(j0 + r) * KPAD + i0 + tx] = tile[tx][r]; // coalesced
    }
}

// ---------------- per-step elementwise kernels ----------------

// fx = tanh(v) (f32, 32x2000) and fxh (f16, 32x2048 zero-padded)
__global__ __launch_bounds__(256) void k_fx(const float* __restrict__ v,
                                            float* __restrict__ fx,
                                            _Float16* __restrict__ fxh) {
    int idx = blockIdx.x * 256 + threadIdx.x;  // over 32*2048
    int b = idx >> 11, j = idx & (KPAD - 1);
    float t = 0.f;
    if (j < NV) { t = tanhf(v[b * NV + j]); fx[b * NV + j] = t; }
    fxh[idx] = (_Float16)t;
}

// err = v - pred (f32) and errh (f16 padded)
__global__ __launch_bounds__(256) void k_err(const float* __restrict__ v,
                                             const float* __restrict__ pred,
                                             float* __restrict__ err,
                                             _Float16* __restrict__ errh) {
    int idx = blockIdx.x * 256 + threadIdx.x;  // over 32*2048
    int b = idx >> 11, j = idx & (KPAD - 1);
    float t = 0.f;
    if (j < NV) { t = v[b * NV + j] - pred[b * NV + j]; err[b * NV + j] = t; }
    errh[idx] = (_Float16)t;
}

// v -= lr * (err - (1-fx^2)*back) on non-sensory vertices (mask == j>=NSENS,
// exactly how the reference constructs update_mask)
__global__ __launch_bounds__(256) void k_update(float* __restrict__ v,
                                                const float* __restrict__ err,
                                                const float* __restrict__ fx,
                                                const float* __restrict__ back) {
    int idx = blockIdx.x * 256 + threadIdx.x;  // exactly 32*2000 = 250 blocks
    int j = idx % NV;
    if (j >= NSENS) {
        float f  = fx[idx];
        float fp = 1.0f - f * f;
        float dv = err[idx] - fp * back[idx];
        v[idx] -= LRV * dv;
    }
}

// ---------------- WMMA GEMM ----------------
// C(32 x 2000, f32) = A(32 x 2048, f16 row-major) x B, where Bn is the N x K
// row-major view of B (i.e. row n of Bn is column n of B). One wave per
// 16-wide N tile; it computes both M tiles (batch = 32) reusing the B fragment.
//
// Fragment layout per CDNA5 ISA (16-bit A 16x32): lane L holds
// row L%16; halves 0..7 -> K+0..7, halves 8..15 -> K+16..23, lanes>=16 add 8.
// B uses the identical per-lane pattern on the N x K view.
__global__ __launch_bounds__(256) void k_gemm(const _Float16* __restrict__ A,
                                              const _Float16* __restrict__ Bn,
                                              float* __restrict__ C) {
    const int K = KPAD;
    int wave = blockIdx.x * 8 + (threadIdx.x >> 5);   // wave32
    int lane = threadIdx.x & 31;
    if (wave >= NV / 16) return;                      // wave-uniform; EXEC all-1 for WMMA
    int n0   = wave * 16;
    int l16  = lane & 15;
    int koff = (lane >> 4) * 8;                       // lanes 16..31: K+8
    const _Float16* a0 = A  + (size_t)(0  + l16) * K + koff;
    const _Float16* a1 = A  + (size_t)(16 + l16) * K + koff;
    const _Float16* bp = Bn + (size_t)(n0 + l16) * K + koff;
    v8f acc0 = {}; v8f acc1 = {};
    for (int k0 = 0; k0 < K; k0 += 32) {
        v8h al0 = *(const v8h*)(a0 + k0);
        v8h ah0 = *(const v8h*)(a0 + k0 + 16);
        v8h al1 = *(const v8h*)(a1 + k0);
        v8h ah1 = *(const v8h*)(a1 + k0 + 16);
        v8h bl  = *(const v8h*)(bp + k0);
        v8h bh  = *(const v8h*)(bp + k0 + 16);
        v16h av0 = __builtin_shufflevector(al0, ah0, 0,1,2,3,4,5,6,7,8,9,10,11,12,13,14,15);
        v16h av1 = __builtin_shufflevector(al1, ah1, 0,1,2,3,4,5,6,7,8,9,10,11,12,13,14,15);
        v16h bv  = __builtin_shufflevector(bl,  bh,  0,1,2,3,4,5,6,7,8,9,10,11,12,13,14,15);
        acc0 = __builtin_amdgcn_wmma_f32_16x16x32_f16(false, av0, false, bv,
                                                      (short)0, acc0, false, false);
        acc1 = __builtin_amdgcn_wmma_f32_16x16x32_f16(false, av1, false, bv,
                                                      (short)0, acc1, false, false);
    }
    // C/D layout: lane%16 = N column; lanes>=16 carry M rows 8..15; VGPR r = M%8
    int n = n0 + l16;
    int mbase = (lane >> 4) * 8;
    #pragma unroll
    for (int r = 0; r < 8; ++r) {
        C[(size_t)(mbase + r) * NV + n]      = acc0[r];
        C[(size_t)(16 + mbase + r) * NV + n] = acc1[r];
    }
}

// ---------------- host launcher ----------------
extern "C" void kernel_launch(void* const* d_in, const int* in_sizes, int n_in,
                              void* d_out, int out_size, void* d_ws, size_t ws_size,
                              hipStream_t stream) {
    const float* values  = (const float*)d_in[0];   // 32 x 2000 f32
    const float* weights = (const float*)d_in[1];   // nE f32
    const int*   eidx    = (const int*)  d_in[2];   // 2 x nE int32
    const int nE = in_sizes[2] / 2;
    const int* srcI = eidx;
    const int* dstI = eidx + nE;

    // workspace layout
    char* ws = (char*)d_ws;
    _Float16* Wh   = (_Float16*)(ws);                         //  8 MB: 2048x2048 f16 (W, row=dst)
    _Float16* WhT  = (_Float16*)(ws + (8u  << 20));           //  8 MB: transpose (row=src)
    float*    Wf   = (float*)   (ws + (16u << 20));           // 16 MB: 2048x2048 f32 accumulator
    _Float16* fxh  = (_Float16*)(ws + (32u << 20));           // 128 KB: 32x2048 f16
    _Float16* errh = (_Float16*)(ws + (32u << 20) + 131072);  // 128 KB
    float*    fx   = (float*)   (ws + (32u << 20) + 262144);  // 250 KB each below
    float*    pred = fx   + NB * NV;
    float*    err  = pred + NB * NV;
    float*    back = err  + NB * NV;

    float* v = (float*)d_out;  // work in-place in d_out (32 x 2000 f32)
    hipMemcpyAsync(v, values, (size_t)NB * NV * sizeof(float),
                   hipMemcpyDeviceToDevice, stream);

    // ---- densify sparse graph once per call ----
    k_zero4<<<(KPAD * KPAD) / 4 / 256, 256, 0, stream>>>((float4*)Wf);
    k_scatter<<<(nE + 255) / 256, 256, 0, stream>>>(srcI, dstI, weights, Wf, nE);
    k_convert<<<(KPAD / 32) * (KPAD / 32), 256, 0, stream>>>(Wf, Wh, WhT);

    // ---- T predictive-coding steps as dense WMMA GEMMs ----
    const int ewBlocks   = (NB * KPAD) / 256;   // 256
    const int updBlocks  = (NB * NV)   / 256;   // 250
    const int gemmBlocks = (NV / 16 + 7) / 8;   // 16 (125 waves, 8 waves/block)
    for (int t = 0; t < TSTEPS; ++t) {
        k_fx  <<<ewBlocks,   256, 0, stream>>>(v, fx, fxh);
        k_gemm<<<gemmBlocks, 256, 0, stream>>>(fxh, Wh, pred);    // pred = fx . W^T
        k_err <<<ewBlocks,   256, 0, stream>>>(v, pred, err, errh);
        k_gemm<<<gemmBlocks, 256, 0, stream>>>(errh, WhT, back);  // back = err . W
        k_update<<<updBlocks, 256, 0, stream>>>(v, err, fx, back);
    }
}
